// DeepSeekMoE_91336774516862
// MI455X (gfx1250) — matrix operand order repaired and checked by reading, not compiled, not run on hardware
//
#include <hip/hip_runtime.h>
#include <hip/hip_bf16.h>
#include <math.h>

#define T_TOK 4096
#define H_DIM 1024
#define F_DIM 4096
#define E_NUM 8

typedef __attribute__((ext_vector_type(16))) __bf16 v16bf;
typedef __attribute__((ext_vector_type(8)))  float  v8f;
typedef __attribute__((ext_vector_type(4)))  int    v4i;

#if __has_builtin(__builtin_amdgcn_global_load_async_to_lds_b128)
#define HAVE_ASYNC_LDS 1
#else
#define HAVE_ASYNC_LDS 0
#endif

#if __has_builtin(__builtin_amdgcn_s_wait_asynccnt)
#define WAIT_ASYNC(n) __builtin_amdgcn_s_wait_asynccnt(n)
#else
#define WAIT_ASYNC(n) asm volatile("s_wait_asynccnt %0" ::"i"(n))
#endif

// Builtin signature (from clang diagnostic): param0 = int4 AS1*, param1 = int4 AS3*
#define AS1CAST(p) ((__attribute__((address_space(1))) v4i*)(p))
#define AS3CAST(p) ((__attribute__((address_space(3))) v4i*)(p))

static __device__ __forceinline__ unsigned short f32_bf16(float f) {
  __bf16 b = (__bf16)f;               // RNE fptrunc
  return __builtin_bit_cast(unsigned short, b);
}
static __device__ __forceinline__ float gelu_exact(float v) {
  return 0.5f * v * (1.0f + erff(v * 0.70710678118654752f));
}

// ---------------------------------------------------------------- conversion
__global__ void __launch_bounds__(256)
cvt_bf16_kernel(const float* __restrict__ src, unsigned short* __restrict__ dst,
                size_t n4) {
  size_t i = (size_t)blockIdx.x * blockDim.x + threadIdx.x;
  const size_t stride = (size_t)gridDim.x * blockDim.x;
  for (; i < n4; i += stride) {
    float4 v = ((const float4*)src)[i];
    ushort4 o;
    o.x = f32_bf16(v.x); o.y = f32_bf16(v.y);
    o.z = f32_bf16(v.z); o.w = f32_bf16(v.w);
    ((ushort4*)dst)[i] = o;
  }
}

// ---------------------------------------------------------------- gating
__global__ void __launch_bounds__(256)
gate_kernel(const float* __restrict__ x, const float* __restrict__ gw,
            const float* __restrict__ gb, const float* __restrict__ rb,
            int* __restrict__ cnt, int* __restrict__ gidx,
            int* __restrict__ slot_e, int* __restrict__ slot_p,
            float* __restrict__ slot_w) {
  const int t    = (int)(((size_t)blockIdx.x * blockDim.x + threadIdx.x) >> 5);
  const int lane = threadIdx.x & 31;
  if (t >= T_TOK) return;
  const float* xr = x + (size_t)t * H_DIM;
  float s[E_NUM];
#pragma unroll
  for (int e = 0; e < E_NUM; ++e) s[e] = 0.f;
  for (int j = lane; j < H_DIM; j += 32) {
    const float xv = xr[j];
#pragma unroll
    for (int e = 0; e < E_NUM; ++e) s[e] += xv * gw[e * H_DIM + j];
  }
#pragma unroll
  for (int e = 0; e < E_NUM; ++e)
    for (int off = 16; off > 0; off >>= 1)
      s[e] += __shfl_xor(s[e], off, 32);
  if (lane == 0) {
    float sc[E_NUM];
#pragma unroll
    for (int e = 0; e < E_NUM; ++e)
      sc[e] = 1.0f / (1.0f + expf(-(s[e] + gb[e] + rb[e])));
    int e0 = 0;
#pragma unroll
    for (int e = 1; e < E_NUM; ++e) if (sc[e] > sc[e0]) e0 = e;
    int e1 = (e0 == 0) ? 1 : 0;
#pragma unroll
    for (int e = 0; e < E_NUM; ++e)
      if (e != e0 && e != e1 && sc[e] > sc[e1]) e1 = e;
    const int es[2] = {e0, e1};
#pragma unroll
    for (int k = 0; k < 2; ++k) {
      const int e = es[k];
      const int p = atomicAdd(&cnt[e], 1);
      gidx[e * T_TOK + p] = t;
      slot_e[t * 2 + k] = e;
      slot_p[t * 2 + k] = p;
      slot_w[t * 2 + k] = sc[e];
    }
  }
}

__global__ void scan_kernel(const int* __restrict__ cnt, int* __restrict__ hbase) {
  if (threadIdx.x == 0 && blockIdx.x == 0) {
    int s = 0;
    for (int e = 0; e < E_NUM; ++e) { hbase[e] = s; s += cnt[e]; }
  }
}

// ---------------------------------------------------------------- WMMA GEMM
// D[M,N] = A[M,K](bf16) * B[N,K](bf16, row-major "out x in") + bias
// Tile: 128x128x32. 8 waves; each wave = 64x32 => 4x2 wmma_f32_16x16x32_bf16.
// Double-buffered LDS in fragment order (each lane's 16-half fragment is one
// contiguous 32-byte run); staged with GLOBAL_LOAD_ASYNC_TO_LDS_B128 and
// s_wait_asynccnt when available.
// MODE 0: v = gelu(v+bias) -> bf16 store. MODE 1: v = v+bias -> f32 store.
template <int MODE>
__global__ void __launch_bounds__(256)
ffn_gemm_kernel(const unsigned short* __restrict__ A,
                const unsigned short* __restrict__ B,
                const float* __restrict__ bias, void* __restrict__ Out,
                int K, int N, size_t BstrideE, int biasStrideE,
                const int* __restrict__ cnt, const int* __restrict__ hbase,
                const int* __restrict__ gidx, int Mcap, int compactA,
                int outCompact) {
  const int e = blockIdx.z;
  const int mcount = cnt ? cnt[e] : Mcap;
  if ((int)blockIdx.x * 128 >= mcount) return;  // routed early-out (whole block)
  const int base = hbase ? hbase[e] : 0;
  const unsigned short* Be = B + (size_t)e * BstrideE;
  const float* be = bias + (size_t)e * (size_t)biasStrideE;

  __shared__ alignas(32) uint4 As4[2][8][32][2];  // 16 KB
  __shared__ alignas(32) uint4 Bs4[2][8][32][2];  // 16 KB

  const int tid  = threadIdx.x;
  const int lane = tid & 31;
  const int wave = tid >> 5;
  const int wm   = wave >> 2;  // 0..1 -> M half
  const int wn   = wave & 3;   // 0..3 -> N quarter
  const int sf   = wave;       // staged fragment index 0..7
  const int l15  = lane & 15;
  const int hi   = (lane & 16) ? 1 : 0;
  const int hi8  = hi * 8;

  // Per-ISA A fragment K layout: lane<16 -> K {0..7,16..23}; lane>=16 -> {8..15,24..31}
  const int kA0 = hi ? 8 : 0;
  // B fragment (same K layout as A): lane<16 -> K {0..7,16..23}; lane>=16 -> {8..15,24..31}
  const int kB0 = hi ? 8 : 0;

  // A row this thread stages (M index within routed space)
  const int m_st = blockIdx.x * 128 + sf * 16 + l15;
  long long arow;
  if (gidx)          arow = (m_st < mcount) ? (long long)gidx[(size_t)e * Mcap + m_st] : -1;
  else if (compactA) arow = (m_st < mcount) ? (long long)(base + m_st) : -1;
  else               arow = m_st;
  // Invalid rows: clamp to row 0 (harmless reads; their D rows are masked at
  // the epilogue store, and a garbage A row only affects its own D row).
  const size_t arowc = (size_t)(arow < 0 ? 0 : arow);
  const int n_st = blockIdx.y * 128 + sf * 16 + l15;

  const v8f vzero = {};
  v8f acc[4][2];
#pragma unroll
  for (int i = 0; i < 4; ++i)
#pragma unroll
    for (int j = 0; j < 2; ++j) acc[i][j] = vzero;

  auto compute_tile = [&](int cur) {
    v16bf aF[4], bF[2];
#pragma unroll
    for (int i = 0; i < 4; ++i)
      aF[i] = *(const v16bf*)&As4[cur][wm * 4 + i][lane][0];
#pragma unroll
    for (int j = 0; j < 2; ++j)
      bF[j] = *(const v16bf*)&Bs4[cur][wn * 2 + j][lane][0];
#pragma unroll
    for (int i = 0; i < 4; ++i)
#pragma unroll
      for (int j = 0; j < 2; ++j)
        acc[i][j] = __builtin_amdgcn_wmma_f32_16x16x32_bf16(
            false, aF[i], false, bF[j], (short)0, acc[i][j], false, false);
  };

#if HAVE_ASYNC_LDS
  auto issue_tile = [&](int k0, int buf) {
    const unsigned short* pa = A + arowc * (size_t)K + k0 + kA0;
    const unsigned short* pb = Be + (size_t)n_st * (size_t)K + k0 + kB0;
    __builtin_amdgcn_global_load_async_to_lds_b128(
        AS1CAST(pa), AS3CAST(&As4[buf][sf][lane][0]), 0, 0);
    __builtin_amdgcn_global_load_async_to_lds_b128(
        AS1CAST(pa + 16), AS3CAST(&As4[buf][sf][lane][1]), 0, 0);
    __builtin_amdgcn_global_load_async_to_lds_b128(
        AS1CAST(pb), AS3CAST(&Bs4[buf][sf][lane][0]), 0, 0);
    __builtin_amdgcn_global_load_async_to_lds_b128(
        AS1CAST(pb + 16), AS3CAST(&Bs4[buf][sf][lane][1]), 0, 0);
  };

  issue_tile(0, 0);
  const int nt = K >> 5;
  for (int i = 0; i < nt; ++i) {
    const int cur = i & 1;
    if (i + 1 < nt) {
      issue_tile((i + 1) << 5, (i + 1) & 1);
      WAIT_ASYNC(4);  // tile i's 4 async loads complete (in-order)
    } else {
      WAIT_ASYNC(0);
    }
    __syncthreads();       // all waves' tile-i data visible in LDS
    compute_tile(cur);
    __syncthreads();       // everyone done reading buf[cur] before reuse
  }
#else
  for (int k0 = 0; k0 < K; k0 += 32) {
    {
      const unsigned short* pa = A + arowc * (size_t)K + k0 + kA0;
      As4[0][sf][lane][0] = *(const uint4*)(pa);
      As4[0][sf][lane][1] = *(const uint4*)(pa + 16);
      const unsigned short* pb = Be + (size_t)n_st * (size_t)K + k0 + kB0;
      Bs4[0][sf][lane][0] = *(const uint4*)(pb);
      Bs4[0][sf][lane][1] = *(const uint4*)(pb + 16);
    }
    __syncthreads();
    compute_tile(0);
    __syncthreads();
  }
#endif

  // ---- epilogue: C/D layout = VGPR r, lane: M=r+(lane<16?0:8), N=lane%16 ----
#pragma unroll
  for (int j = 0; j < 2; ++j) {
    const int ng = blockIdx.y * 128 + (wn * 2 + j) * 16 + l15;
    const float bv = be[ng];
#pragma unroll
    for (int i = 0; i < 4; ++i) {
      const int mb = blockIdx.x * 128 + (wm * 4 + i) * 16 + hi8;
#pragma unroll
      for (int r = 0; r < 8; ++r) {
        const int m = mb + r;
        if (m < mcount) {
          float v = acc[i][j][r] + bv;
          const size_t orow = outCompact ? (size_t)(base + m) : (size_t)m;
          if (MODE == 0) {
            ((unsigned short*)Out)[orow * (size_t)N + ng] =
                f32_bf16(gelu_exact(v));
          } else {
            ((float*)Out)[orow * (size_t)N + ng] = v;
          }
        }
      }
    }
  }
}

// ---------------------------------------------------------------- combine
__global__ void __launch_bounds__(256)
combine_kernel(float* __restrict__ out, const float* __restrict__ yc,
               const int* __restrict__ slot_e, const int* __restrict__ slot_p,
               const float* __restrict__ slot_w, const int* __restrict__ hbase) {
  const size_t gid = (size_t)blockIdx.x * blockDim.x + threadIdx.x;
  const int t = (int)(gid / (H_DIM / 4));
  const int c = (int)(gid % (H_DIM / 4)) * 4;
  if (t >= T_TOK) return;
  const int e0 = slot_e[t * 2 + 0], p0 = slot_p[t * 2 + 0];
  const int e1 = slot_e[t * 2 + 1], p1 = slot_p[t * 2 + 1];
  const float w0 = slot_w[t * 2 + 0], w1 = slot_w[t * 2 + 1];
  const size_t r0 = ((size_t)(hbase[e0] + p0)) * H_DIM + c;
  const size_t r1 = ((size_t)(hbase[e1] + p1)) * H_DIM + c;
  float4 o = *(float4*)&out[(size_t)t * H_DIM + c];
  const float4 y0 = *(const float4*)&yc[r0];
  const float4 y1 = *(const float4*)&yc[r1];
  o.x += w0 * y0.x + w1 * y1.x;
  o.y += w0 * y0.y + w1 * y1.y;
  o.z += w0 * y0.z + w1 * y1.z;
  o.w += w0 * y0.w + w1 * y1.w;
  *(float4*)&out[(size_t)t * H_DIM + c] = o;
}

// ---------------------------------------------------------------- launch
extern "C" void kernel_launch(void* const* d_in, const int* in_sizes, int n_in,
                              void* d_out, int out_size, void* d_ws,
                              size_t ws_size, hipStream_t stream) {
  (void)in_sizes; (void)n_in; (void)out_size; (void)ws_size;
  const float* x   = (const float*)d_in[0];
  const float* gw  = (const float*)d_in[1];
  const float* gb  = (const float*)d_in[2];
  const float* rb  = (const float*)d_in[3];
  const float* sw1 = (const float*)d_in[4];
  const float* sb1 = (const float*)d_in[5];
  const float* sw2 = (const float*)d_in[6];
  const float* sb2 = (const float*)d_in[7];
  const float* ew1 = (const float*)d_in[8];
  const float* eb1 = (const float*)d_in[9];
  const float* ew2 = (const float*)d_in[10];
  const float* eb2 = (const float*)d_in[11];
  float* out = (float*)d_out;

  char* ws = (char*)d_ws;
  size_t off = 0;
  auto take = [&](size_t bytes) -> char* {
    char* p = ws + off;
    off += bytes;
    off = (off + 255) & ~(size_t)255;
    return p;
  };
  unsigned short* xh   = (unsigned short*)take((size_t)T_TOK * H_DIM * 2);
  unsigned short* w1s  = (unsigned short*)take((size_t)F_DIM * H_DIM * 2);
  unsigned short* w2s  = (unsigned short*)take((size_t)H_DIM * F_DIM * 2);
  unsigned short* w1e  = (unsigned short*)take((size_t)E_NUM * F_DIM * H_DIM * 2);
  unsigned short* w2e  = (unsigned short*)take((size_t)E_NUM * H_DIM * F_DIM * 2);
  unsigned short* hbuf = (unsigned short*)take((size_t)2 * T_TOK * F_DIM * 2);
  float* yc    = (float*)take((size_t)2 * T_TOK * H_DIM * 4);
  int* cnt     = (int*)take(E_NUM * 4);
  int* hbase   = (int*)take(E_NUM * 4);
  int* gidx    = (int*)take((size_t)E_NUM * T_TOK * 4);
  int* slot_e  = (int*)take((size_t)2 * T_TOK * 4);
  int* slot_p  = (int*)take((size_t)2 * T_TOK * 4);
  float* slot_w = (float*)take((size_t)2 * T_TOK * 4);

  (void)hipMemsetAsync(cnt, 0, E_NUM * sizeof(int), stream);

  gate_kernel<<<T_TOK / 8, 256, 0, stream>>>(x, gw, gb, rb, cnt, gidx, slot_e,
                                             slot_p, slot_w);
  scan_kernel<<<1, 32, 0, stream>>>(cnt, hbase);

  cvt_bf16_kernel<<<2048, 256, 0, stream>>>(x, xh, (size_t)T_TOK * H_DIM / 4);
  cvt_bf16_kernel<<<2048, 256, 0, stream>>>(sw1, w1s, (size_t)F_DIM * H_DIM / 4);
  cvt_bf16_kernel<<<2048, 256, 0, stream>>>(sw2, w2s, (size_t)H_DIM * F_DIM / 4);
  cvt_bf16_kernel<<<8192, 256, 0, stream>>>(ew1, w1e,
                                            (size_t)E_NUM * F_DIM * H_DIM / 4);
  cvt_bf16_kernel<<<8192, 256, 0, stream>>>(ew2, w2e,
                                            (size_t)E_NUM * H_DIM * F_DIM / 4);

  // shared FFN: h = gelu(x @ w1^T + b1); out = h @ w2^T + b2
  ffn_gemm_kernel<0><<<dim3(32, 32, 1), 256, 0, stream>>>(
      xh, w1s, sb1, hbuf, H_DIM, F_DIM, 0, 0, nullptr, nullptr, nullptr, T_TOK,
      0, 0);
  ffn_gemm_kernel<1><<<dim3(32, 8, 1), 256, 0, stream>>>(
      hbuf, w2s, sb2, out, F_DIM, H_DIM, 0, 0, nullptr, nullptr, nullptr,
      T_TOK, 0, 0);

  // routed experts on compacted token lists
  ffn_gemm_kernel<0><<<dim3(32, 32, E_NUM), 256, 0, stream>>>(
      xh, w1e, eb1, hbuf, H_DIM, F_DIM, (size_t)F_DIM * H_DIM, F_DIM, cnt,
      hbase, gidx, T_TOK, 0, 1);
  ffn_gemm_kernel<1><<<dim3(32, 8, E_NUM), 256, 0, stream>>>(
      hbuf, w2e, eb2, yc, F_DIM, H_DIM, (size_t)H_DIM * F_DIM, H_DIM, cnt,
      hbase, nullptr, T_TOK, 1, 1);

  combine_kernel<<<(T_TOK * (H_DIM / 4)) / 256, 256, 0, stream>>>(
      out, yc, slot_e, slot_p, slot_w, hbase);
}
